// LSTM_7052336300356
// MI455X (gfx1250) — matrix operand order repaired and checked by reading, not compiled, not run on hardware
//
#include <hip/hip_runtime.h>
#include <hip/hip_bf16.h>

// ---------------------------------------------------------------------------
// 2-layer LSTM (H=1024, B=512, T=256, F=32) for gfx1250 (MI455X).
// Per-timestep fused GEMM+gate kernels using v_wmma_f32_16x16x32_bf16.
// Weights converted fp32->bf16 once per launch; they stay L2-resident (24 MB
// bf16 << 192 MB L2), so the problem is matrix-pipe / dependency bound.
// Wave tile: 32x64 (8 WMMAs per 12 b128 loads per K-step).
// ---------------------------------------------------------------------------

#define HID   1024
#define BATCH 512
#define TSTEP 256
#define FPRED 32

typedef unsigned short u16;
typedef __attribute__((ext_vector_type(16))) __bf16        v16bf;
typedef __attribute__((ext_vector_type(8)))  float         v8f;
typedef __attribute__((ext_vector_type(8)))  unsigned int  v8u;

// fp32 -> bf16 (round to nearest even)
__device__ __forceinline__ u16 f2bf(float f) {
  unsigned int u = __float_as_uint(f);
  unsigned int r = u + 0x7FFFu + ((u >> 16) & 1u);
  return (u16)(r >> 16);
}
__device__ __forceinline__ float bf2f(u16 b) {
  return __uint_as_float(((unsigned int)b) << 16);
}
__device__ __forceinline__ float sigmoidf(float x) {
  return 1.0f / (1.0f + __expf(-x));
}

// A-fragment (16x32 bf16, MxK). ISA 7.12.2: lanes 0-15 -> rows, half=lane>>4.
// K order used here (as B): dwords 0-3: K = kb + 16*half .. +7 ; dwords 4-7: K = kb + 16*half + 8 .. +7
__device__ __forceinline__ v16bf load_a_frag(const u16* __restrict__ A,
                                             int row, int kb, int half) {
  const uint4* p0 = reinterpret_cast<const uint4*>(A + (size_t)row * HID + kb + half * 16);
  const uint4* p1 = reinterpret_cast<const uint4*>(A + (size_t)row * HID + kb + half * 16 + 8);
  uint4 x = *p0, y = *p1;
  v8u u;
  u[0] = x.x; u[1] = x.y; u[2] = x.z; u[3] = x.w;
  u[4] = y.x; u[5] = y.y; u[6] = y.z; u[7] = y.w;
  return __builtin_bit_cast(v16bf, u);
}

// B-fragment (32x16 bf16, KxN). gates = h @ W^T so B(k,n) = W[n,k]:
// lane (n = lane&15) holds 16 contiguous K values starting at kb + 16*half.
__device__ __forceinline__ v16bf load_b_frag(const u16* __restrict__ W,
                                             int n, int kb, int half) {
  const uint4* p = reinterpret_cast<const uint4*>(W + (size_t)n * HID + kb + half * 16);
  uint4 x = p[0], y = p[1];
  v8u u;
  u[0] = x.x; u[1] = x.y; u[2] = x.z; u[3] = x.w;
  u[4] = y.x; u[5] = y.y; u[6] = y.z; u[7] = y.w;
  return __builtin_bit_cast(v16bf, u);
}

__device__ __forceinline__ v8f wmma_bf16(v16bf a, v16bf b, v8f c) {
  return __builtin_amdgcn_wmma_f32_16x16x32_bf16(
      false, a, false, b, (short)0, c, false, false);
}

// Accumulate a 32x64 wave tile: acc += A[rbase:+32, :] @ W[nbase:+64, :]^T
__device__ __forceinline__ void gemm_tile(v8f acc[2][4],
                                          const u16* __restrict__ A,
                                          const u16* __restrict__ W,
                                          int rbase, int nbase,
                                          int half, int l16) {
#pragma unroll 4
  for (int kb = 0; kb < HID; kb += 32) {
    v16bf a0 = load_a_frag(A, rbase + l16,      kb, half);
    v16bf a1 = load_a_frag(A, rbase + 16 + l16, kb, half);
    v16bf b0 = load_b_frag(W, nbase +      l16, kb, half);
    v16bf b1 = load_b_frag(W, nbase + 16 + l16, kb, half);
    v16bf b2 = load_b_frag(W, nbase + 32 + l16, kb, half);
    v16bf b3 = load_b_frag(W, nbase + 48 + l16, kb, half);
    acc[0][0] = wmma_bf16(a0, b0, acc[0][0]);
    acc[0][1] = wmma_bf16(a0, b1, acc[0][1]);
    acc[0][2] = wmma_bf16(a0, b2, acc[0][2]);
    acc[0][3] = wmma_bf16(a0, b3, acc[0][3]);
    acc[1][0] = wmma_bf16(a1, b0, acc[1][0]);
    acc[1][1] = wmma_bf16(a1, b1, acc[1][1]);
    acc[1][2] = wmma_bf16(a1, b2, acc[1][2]);
    acc[1][3] = wmma_bf16(a1, b3, acc[1][3]);
  }
}

// ---------------------------------------------------------------------------
// Fused LSTM cell: gates = A1@W1^T (+ A2@W2^T) + b_ih + b_hh (+ x*wx for L0)
// Workgroup: 64 rows (batch) x 64 cols (j within H), 4 gates looped i,f,g,o.
// Block = 64 threads = 2 waves; each wave owns 32 rows x 64 cols.
// ---------------------------------------------------------------------------
template <bool LAYER1>
__global__ __launch_bounds__(64)
void lstm_cell_kernel(const u16* __restrict__ A1, const u16* __restrict__ W1,
                      const u16* __restrict__ A2, const u16* __restrict__ W2,
                      const float* __restrict__ xvec, const float* __restrict__ wx,
                      const float* __restrict__ bih, const float* __restrict__ bhh,
                      float* __restrict__ c, u16* __restrict__ h_out) {
  __shared__ float s_acc[64 * 64];  // current gate pre-activation tile
  __shared__ float s_t[64 * 64];    // sigmoid(i), later tanh(c_new)
  __shared__ float s_c[64 * 64];    // sigmoid(f) * c_old

  const int tid  = threadIdx.x;      // 0..63
  const int lane = tid & 31;
  const int w    = tid >> 5;         // wave 0/1 -> rows w*32..w*32+31
  const int half = lane >> 4;
  const int l16  = lane & 15;
  const int m0   = blockIdx.y * 64;  // batch-row tile
  const int j0   = blockIdx.x * 64;  // hidden-col tile

  const int rbase = m0 + w * 32;

#pragma unroll 1
  for (int g = 0; g < 4; ++g) {      // PyTorch gate order: i, f, g, o
    v8f z = {0.f, 0.f, 0.f, 0.f, 0.f, 0.f, 0.f, 0.f};
    v8f acc[2][4] = {{z, z, z, z}, {z, z, z, z}};
    const int nbase = g * HID + j0;

    gemm_tile(acc, A1, W1, rbase, nbase, half, l16);
    if (LAYER1) gemm_tile(acc, A2, W2, rbase, nbase, half, l16);

    __syncthreads();  // WAR: previous gate epilogue finished reading s_acc
#pragma unroll
    for (int mi = 0; mi < 2; ++mi)
#pragma unroll
      for (int ni = 0; ni < 4; ++ni)
#pragma unroll
        for (int r = 0; r < 8; ++r) {
          // C/D layout: VGPR r -> row r (lanes 0-15) / r+8 (lanes 16-31)
          int rl = w * 32 + mi * 16 + half * 8 + r;
          int cl = ni * 16 + l16;
          s_acc[rl * 64 + cl] = acc[mi][ni][r];
        }
    __syncthreads();  // RAW: epilogue reads s_acc

    // Fused gate epilogue: thread tid owns column (j0+tid) for all 64 rows,
    // so per-gate column constants hoist out of the row loop.
    {
      const int n    = g * HID + j0 + tid;
      const float bs = bih[n] + bhh[n];
      const float wn = LAYER1 ? 0.0f : wx[n];
#pragma unroll 4
      for (int e = 0; e < 64; ++e) {
        int idx = e * 64 + tid;            // row e, col tid
        float pre = s_acc[idx] + bs;
        if (!LAYER1) pre += xvec[m0 + e] * wn;   // rank-1 input projection
        size_t gi = (size_t)(m0 + e) * HID + j0 + tid;
        if (g == 0) {
          s_t[idx] = sigmoidf(pre);                      // sig(i)
        } else if (g == 1) {
          s_c[idx] = sigmoidf(pre) * c[gi];              // sig(f)*c_old
        } else if (g == 2) {
          float cn = s_c[idx] + s_t[idx] * tanhf(pre);   // c_new
          c[gi]    = cn;
          s_t[idx] = tanhf(cn);
        } else {
          float hv = sigmoidf(pre) * s_t[idx];           // h_new
          h_out[gi] = f2bf(hv);
        }
      }
    }
  }
}

// out[b] = h1[b,:] . W_lin + b_lin  (one wave per batch row)
__global__ __launch_bounds__(128)
void linear_kernel(const u16* __restrict__ h1, const float* __restrict__ Wlin,
                   const float* __restrict__ blin, float* __restrict__ out) {
  int row  = blockIdx.x * 4 + (threadIdx.x >> 5);
  int lane = threadIdx.x & 31;
  const u16* hr = h1 + (size_t)row * HID;
  float s = 0.f;
#pragma unroll 4
  for (int j = lane; j < HID; j += 32) s += bf2f(hr[j]) * Wlin[j];
#pragma unroll
  for (int off = 16; off > 0; off >>= 1) s += __shfl_xor(s, off, 32);
  if (lane == 0) out[row] = s + blin[0];
}

__global__ void cvt_f32_bf16_kernel(const float* __restrict__ in,
                                    u16* __restrict__ out, int n) {
  int i = blockIdx.x * 256 + threadIdx.x;
  if (i < n) out[i] = f2bf(in[i]);
}

// ---------------------------------------------------------------------------
extern "C" void kernel_launch(void* const* d_in, const int* in_sizes, int n_in,
                              void* d_out, int out_size, void* d_ws, size_t ws_size,
                              hipStream_t stream) {
  (void)in_sizes; (void)n_in; (void)out_size; (void)ws_size;
  const float* input = (const float*)d_in[0];   // [T*B, 1]
  const float* W_ih0 = (const float*)d_in[1];   // [4H, 1]
  const float* W_hh0 = (const float*)d_in[2];   // [4H, H]
  const float* b_ih0 = (const float*)d_in[3];
  const float* b_hh0 = (const float*)d_in[4];
  const float* W_ih1 = (const float*)d_in[5];   // [4H, H]
  const float* W_hh1 = (const float*)d_in[6];   // [4H, H]
  const float* b_ih1 = (const float*)d_in[7];
  const float* b_hh1 = (const float*)d_in[8];
  const float* W_lin = (const float*)d_in[9];   // [1, H]
  const float* b_lin = (const float*)d_in[10];
  float* out = (float*)d_out;                   // [(T+F)*B]

  const size_t WSZ = (size_t)4 * HID * HID;     // 4H*H elements per weight
  const size_t HSZ = (size_t)BATCH * HID;

  u16* Whh0b = (u16*)d_ws;
  u16* Wih1b = Whh0b + WSZ;
  u16* Whh1b = Wih1b + WSZ;
  u16* h0buf = Whh1b + WSZ;                     // h0 ping/pong (bf16)
  u16* h1buf = h0buf + 2 * HSZ;                 // h1 ping/pong (bf16)
  float* c0  = (float*)(h1buf + 2 * HSZ);
  float* c1  = c0 + HSZ;

  // Convert weights fp32 -> bf16 (every call; deterministic)
  {
    int n = (int)WSZ;
    int nb = (n + 255) / 256;
    cvt_f32_bf16_kernel<<<nb, 256, 0, stream>>>(W_hh0, Whh0b, n);
    cvt_f32_bf16_kernel<<<nb, 256, 0, stream>>>(W_ih1, Wih1b, n);
    cvt_f32_bf16_kernel<<<nb, 256, 0, stream>>>(W_hh1, Whh1b, n);
  }
  // Zero initial h0/h1 (bf16 0 == 0x0000) and c0/c1 (fp32 0): contiguous region
  hipMemsetAsync(h0buf, 0, 4 * HSZ * sizeof(u16) + 2 * HSZ * sizeof(float), stream);

  dim3 grid(HID / 64, BATCH / 64);   // 16 x 8 = 128 workgroups
  dim3 block(64);

  int ping = 0;
  for (int t = 0; t < TSTEP + FPRED; ++t) {
    const float* xptr = (t < TSTEP) ? (input + (size_t)t * BATCH)
                                    : (out + (size_t)(t - 1) * BATCH);
    u16* h0p = h0buf + (size_t)ping * HSZ;
    u16* h0n = h0buf + (size_t)(1 - ping) * HSZ;
    u16* h1p = h1buf + (size_t)ping * HSZ;
    u16* h1n = h1buf + (size_t)(1 - ping) * HSZ;

    // Layer 0: gates = h0_prev @ Whh0^T + x*W_ih0 + b_ih0 + b_hh0
    lstm_cell_kernel<false><<<grid, block, 0, stream>>>(
        h0p, Whh0b, nullptr, nullptr, xptr, W_ih0, b_ih0, b_hh0, c0, h0n);

    // Layer 1: gates = h0_new @ Wih1^T + h1_prev @ Whh1^T + b_ih1 + b_hh1
    lstm_cell_kernel<true><<<grid, block, 0, stream>>>(
        h0n, Wih1b, h1p, Whh1b, nullptr, nullptr, b_ih1, b_hh1, c1, h1n);

    // Readout
    linear_kernel<<<BATCH / 4, 128, 0, stream>>>(h1n, W_lin, b_lin,
                                                 out + (size_t)t * BATCH);
    ping ^= 1;
  }
}
